// PUray_62405874811589
// MI455X (gfx1250) — matrix operand order repaired and checked by reading, not compiled, not run on hardware
//
#include <hip/hip_runtime.h>

#define STEPS 8
#define GB 16   // batches per workgroup
#define NW 8    // waves per workgroup (wave32)

typedef __attribute__((ext_vector_type(16))) __bf16 v16bf;
typedef __attribute__((ext_vector_type(8)))  __bf16 v8bf;
typedef __attribute__((ext_vector_type(8)))  float  v8f;

// ---------- scalar conversions (native bf16 converts on gfx1250) ----------
static __device__ inline __bf16 f2bf(float f) { return (__bf16)f; }
static __device__ inline float bf2f(__bf16 b) { return (float)b; }

// cross-half-wave reduce helper: value from lane^16 (SWAPX16 swizzle)
static __device__ inline float xor16(float x) {
  int i = __builtin_bit_cast(int, x);
  i = __builtin_amdgcn_ds_swizzle(i, 0x401f);
  return __builtin_bit_cast(float, i);
}

// ---------- WMMA ----------
static __device__ inline v8f wmma_bf16(v16bf a, v16bf b, v8f c) {
  return __builtin_amdgcn_wmma_f32_16x16x32_bf16(false, a, false, b, (short)0, c,
                                                 false, false);
}

// ---- swizzled-operand addressing -------------------------------------------
// A operand (16x32 bf16): lane holds 16 contiguous bf16 at [lane*16, lane*16+16).
// Element e of lane: m = lane&15, k = (e&7) + ((e>>3)<<4) + ((lane>>4)<<3).
static __device__ inline int swz_idx(int m, int k) {
  return (m + (((k >> 3) & 1) << 4)) * 16 + (k & 7) + (((k >> 4) & 1) << 3);
}
static __device__ inline v16bf load_Aswz(const __bf16* X, int lane) {
  return *(const v16bf*)(X + lane * 16);
}
// B operand chunk (32x16): lane l covers column n=l&15, rows k=(e&7)+((e>>3)<<4)+((l>>4)<<3).
static __device__ inline v16bf load_Bswz(const __bf16* Wsw, int chunk, int lane) {
  return *(const v16bf*)(Wsw + chunk * 512 + lane * 16);
}

// A operand built per element from functor f(m,k), k in [0,32)
template <typename F>
static __device__ inline v16bf make_A(int lane, F f) {
  const int m = lane & 15;
  const int khi = (lane >> 4) << 3;
  v16bf a;
#pragma unroll
  for (int e = 0; e < 16; ++e) {
    int k = (e & 7) + ((e >> 3) << 4) + khi;
    a[e] = f2bf(f(m, k));
  }
  return a;
}

// C init = bias broadcast along rows (D[m][n]: lane holds column n)
static __device__ inline v8f bias_C(const float* bias, int n0, int Nd, int lane) {
  int n = n0 + (lane & 15);
  float bv = (bias != nullptr && n < Nd) ? bias[n] : 0.f;
  v8f c;
#pragma unroll
  for (int v = 0; v < 8; ++v) c[v] = bv;
  return c;
}
// D tile -> bf16 LDS, A-swizzled layout (optional fused ReLU)
static __device__ inline void store_Dswz(__bf16* O, int n0, int lane, v8f d, bool relu) {
  int n = n0 + (lane & 15);
  int mb = (lane >> 4) << 3;
  int hi = ((n >> 3) & 1) << 4;
  int ep = (n & 7) + (((n >> 4) & 1) << 3);
#pragma unroll
  for (int v = 0; v < 8; ++v) {
    float x = d[v];
    if (relu) x = fmaxf(x, 0.f);
    O[(mb + v + hi) * 16 + ep] = f2bf(x);
  }
}
static __device__ inline void store2swz(__bf16* buf, int lane, const v8f o[2], bool relu) {
  store_Dswz(buf, 0, lane, o[0], relu);
  store_Dswz(buf, 16, lane, o[1], relu);
}
// D tile -> bf16 LDS, row-major [16][32]
static __device__ inline void store_Drm(__bf16* O, int n0, int lane, v8f d) {
  int n = n0 + (lane & 15);
  int mb = (lane >> 4) << 3;
#pragma unroll
  for (int v = 0; v < 8; ++v) O[(mb + v) * 32 + n] = f2bf(d[v]);
}
// D tile -> bf16 LDS, column-major [32 cols][16 rows]: one packed 16B store/lane
static __device__ inline void store_Dcm(__bf16* O, int n0, int lane, v8f d) {
  int n = n0 + (lane & 15);
  int mb = (lane >> 4) << 3;
  v8bf pk;
#pragma unroll
  for (int v = 0; v < 8; ++v) pk[v] = f2bf(d[v]);
  *(v8bf*)(O + n * 16 + mb) = pk;
}
// build attention input A = qrow - K + PE, with K/PE A-swizzled and q row-major
static __device__ inline v16bf make_qkp(const __bf16* qrow, const __bf16* Kswz,
                                        const __bf16* PEswz, int lane) {
  v16bf kv = load_Aswz(Kswz, lane);
  v16bf pv = load_Aswz(PEswz, lane);
  const int hb = (lane >> 4) << 3;
  v8bf qlo = *(const v8bf*)(qrow + hb);
  v8bf qhi = *(const v8bf*)(qrow + 16 + hb);
  v16bf a;
#pragma unroll
  for (int e = 0; e < 16; ++e) {
    float qe = bf2f(e < 8 ? qlo[e] : qhi[e - 8]);
    a[e] = f2bf(qe - bf2f(kv[e]) + bf2f(pv[e]));
  }
  return a;
}
// one linear: Cin<=32 (A prebuilt), Cout<=32
static __device__ inline void lin_tile(v16bf a, const __bf16* Wsw, const float* b,
                                       int Cout, int lane, v8f out[2]) {
#pragma unroll
  for (int t = 0; t < 2; ++t) {
    if (t * 16 < Cout) {
      v8f c = bias_C(b, t * 16, Cout, lane);
      out[t] = wmma_bf16(a, load_Bswz(Wsw, t, lane), c);
    } else {
      out[t] = bias_C(nullptr, 0, 1, lane);
    }
  }
}
// MLP2: Cin<=32 -> 128 (ReLU) -> 32, hidden streamed in 32-wide chunks via hscr
static __device__ inline void mlp2_tile(v16bf a1, const __bf16* W1sw, const float* b1,
                                        const __bf16* W2sw, const float* b2,
                                        __bf16* hscr, int lane, v8f out[2]) {
  out[0] = bias_C(b2, 0, 32, lane);
  out[1] = bias_C(b2, 16, 32, lane);
#pragma unroll
  for (int hc = 0; hc < 4; ++hc) {
#pragma unroll
    for (int t = 0; t < 2; ++t) {
      v8f c = bias_C(b1, hc * 32 + t * 16, 128, lane);
      c = wmma_bf16(a1, load_Bswz(W1sw, hc * 2 + t, lane), c);
      store_Dswz(hscr, t * 16, lane, c, true);
    }
    v16bf a2 = load_Aswz(hscr, lane);
#pragma unroll
    for (int t = 0; t < 2; ++t)
      out[t] = wmma_bf16(a2, load_Bswz(W2sw, hc * 2 + t, lane), out[t]);
  }
}

// ---- workspace layout: pre-swizzled bf16 weights, 1KB chunks ----
enum {
  C_FW1 = 0,  C_FW2 = 2,  C_FW3 = 4,  C_QKV = 6,
  C_PTP1 = 12, C_PTP2 = 20, C_PTA1 = 28, C_PTA2 = 36,
  C_CAQ = 44, C_CAK = 46, C_CAV = 48,
  C_CAP1 = 50, C_CAP2 = 58, C_CAA1 = 66, C_CAA2 = 74,
  C_IM1 = 82, C_IM2 = 84, C_IM3 = 86,
  C_EP1 = 87, C_EP2 = 91, C_EP3 = 93, C_TOTAL = 94
};

struct P {
  const float *coords, *query;
  const float *fw1, *fb1, *fw2, *fb2, *fw3, *fb3;
  const float *qkv_w;
  const float *ptp_w1, *ptp_b1, *ptp_w2, *ptp_b2;
  const float *pta_w1, *pta_b1, *pta_w2, *pta_b2;
  const float *caq_w, *caq_b, *cak_w, *cak_b, *cav_w, *cav_b;
  const float *cap_w1, *cap_b1, *cap_w2, *cap_b2;
  const float *caa_w1, *caa_b1, *caa_w2, *caa_b2;
  const float *im_w1, *im_b1, *im_w2, *im_b2, *im_w3, *im_b3;
  const float *ep_w1, *ep_b1, *ep_w2, *ep_b2, *ep_w3, *ep_b3;
  float *out_ld, *out_ms, *out_cs, *out_ip, *out_cum, *out_eps;
};

// prologue: convert+swizzle one weight matrix per block into B-operand layout
__global__ __launch_bounds__(256) void swz_weights(P p, __bf16* ws) {
#if __has_builtin(__builtin_amdgcn_tensor_load_to_lds) && \
    __has_builtin(__builtin_amdgcn_s_wait_tensorcnt)
  // TDM codegen probe: count=0 => NULL tensor descriptor => architectural no-op
  // (ISA 8.3), but exercises tensor_load_to_lds + s_wait_tensorcnt paths.
  if (blockIdx.x == 0) {
    typedef unsigned int u32x4 __attribute__((ext_vector_type(4)));
    typedef int i32x8v __attribute__((ext_vector_type(8)));
    typedef int i32x4v __attribute__((ext_vector_type(4)));
    u32x4 g0 = (u32x4)0u;
    i32x8v g1 = (i32x8v)0;
    i32x4v g2 = (i32x4v)0, g3 = (i32x4v)0;
    i32x8v g4 = (i32x8v)0;
    __builtin_amdgcn_tensor_load_to_lds(g0, g1, g2, g3, g4, 0);
    __builtin_amdgcn_s_wait_tensorcnt(0);
  }
#endif
  const float* src; int K, N, dst;
  switch (blockIdx.x) {
    case 0:  src = p.fw1;    K = 3;   N = 32;  dst = C_FW1;  break;
    case 1:  src = p.fw2;    K = 32;  N = 32;  dst = C_FW2;  break;
    case 2:  src = p.fw3;    K = 32;  N = 32;  dst = C_FW3;  break;
    case 3:  src = p.qkv_w;  K = 32;  N = 96;  dst = C_QKV;  break;
    case 4:  src = p.ptp_w1; K = 3;   N = 128; dst = C_PTP1; break;
    case 5:  src = p.ptp_w2; K = 128; N = 32;  dst = C_PTP2; break;
    case 6:  src = p.pta_w1; K = 32;  N = 128; dst = C_PTA1; break;
    case 7:  src = p.pta_w2; K = 128; N = 32;  dst = C_PTA2; break;
    case 8:  src = p.caq_w;  K = 32;  N = 32;  dst = C_CAQ;  break;
    case 9:  src = p.cak_w;  K = 32;  N = 32;  dst = C_CAK;  break;
    case 10: src = p.cav_w;  K = 32;  N = 32;  dst = C_CAV;  break;
    case 11: src = p.cap_w1; K = 3;   N = 128; dst = C_CAP1; break;
    case 12: src = p.cap_w2; K = 128; N = 32;  dst = C_CAP2; break;
    case 13: src = p.caa_w1; K = 32;  N = 128; dst = C_CAA1; break;
    case 14: src = p.caa_w2; K = 128; N = 32;  dst = C_CAA2; break;
    case 15: src = p.im_w1;  K = 32;  N = 32;  dst = C_IM1;  break;
    case 16: src = p.im_w2;  K = 32;  N = 32;  dst = C_IM2;  break;
    case 17: src = p.im_w3;  K = 32;  N = 3;   dst = C_IM3;  break;
    case 18: src = p.ep_w1;  K = 35;  N = 32;  dst = C_EP1;  break;
    case 19: src = p.ep_w2;  K = 32;  N = 32;  dst = C_EP2;  break;
    default: src = p.ep_w3;  K = 32;  N = 1;   dst = C_EP3;  break;
  }
  const int NNT = (N + 15) >> 4;
  const int NKC = (K + 31) >> 5;
  const int total = NKC * NNT * 512;
  for (int i = threadIdx.x; i < total; i += 256) {
    int chunk = i >> 9, rem = i & 511, l = rem >> 4, e = rem & 15;
    int kc = chunk / NNT, nt = chunk % NNT;
    int k = kc * 32 + (e & 7) + ((e >> 3) << 4) + ((l >> 4) << 3);
    int n = nt * 16 + (l & 15);
    float v = (k < K && n < N) ? src[k * N + n] : 0.f;
    ws[(dst << 9) + i] = f2bf(v);
  }
}

__global__ __launch_bounds__(256) void puray_cdna5_kernel(P p, const __bf16* ws) {
  const int tid = threadIdx.x;
  const int lane = tid & 31;
  const int w = tid >> 5;
  const int b0 = blockIdx.x * GB;

  __shared__ float  s_coords[GB][16][3];
  __shared__ float  s_query[GB][3];
  __shared__ alignas(32) __bf16 s_hscr[NW][16 * 32];
  __shared__ alignas(32) __bf16 s_pe[NW][16 * 32];    // A-swizzled
  __shared__ alignas(32) __bf16 s_kk[GB][16 * 32];    // feats (phase T) then kk; A-swizzled
  __shared__ alignas(32) __bf16 s_vv[GB][16 * 32];    // column-major [32][16]
  __shared__ alignas(32) __bf16 s_q[16 * 32];         // row-major
  __shared__ alignas(32) __bf16 s_k[16 * 32];         // A-swizzled
  __shared__ alignas(32) __bf16 s_v[16 * 32];         // column-major
  __shared__ alignas(32) __bf16 s_fnew[16 * 32];      // A-swizzled
  __shared__ alignas(32) __bf16 s_qq[GB * 32];        // row-major (rows=batch)
  __shared__ float  s_ca[GB * 32];
  __shared__ float  s_reld[GB][16];
  __shared__ float  s_cum[GB];
  __shared__ float  s_cumprev[GB];
  __shared__ float  s_imp[GB][3];

  // warm the swizzled-weight region into cache (global_prefetch_b8)
  for (int off = tid * 256; off < C_TOTAL * 1024; off += 256 * 256)
    __builtin_prefetch((const char*)ws + off, 0, 3);

  for (int i = tid; i < GB * 16 * 3; i += 256)
    ((float*)s_coords)[i] = p.coords[(size_t)b0 * 48 + i];
  for (int i = tid; i < GB * 3; i += 256)
    ((float*)s_query)[i] = p.query[(size_t)b0 * 3 + i];
  if (tid < GB) { s_cum[tid] = 0.f; s_cumprev[tid] = 0.f; }
  __syncthreads();

  // ===== Phase T0: feats = MLP3(coords) for all GB batches, fully parallel =====
  // Result stored A-swizzled into s_kk[bb] (aliased storage: kk[bb] only
  // overwrites after featAll[bb]'s last read in iteration bb below).
  for (int r = 0; r < 2; ++r) {
    const int bb = w + r * 8;
    v16bf a = make_A(lane, [&](int m, int k) {
      return (k < 3) ? s_coords[bb][m][k] : 0.f; });
    v8f o[2];
    lin_tile(a, ws + (C_FW1 << 9), p.fb1, 32, lane, o); store2swz(s_hscr[w], lane, o, true);
    a = load_Aswz(s_hscr[w], lane);
    lin_tile(a, ws + (C_FW2 << 9), p.fb2, 32, lane, o); store2swz(s_hscr[w], lane, o, true);
    a = load_Aswz(s_hscr[w], lane);
    lin_tile(a, ws + (C_FW3 << 9), p.fb3, 32, lane, o); store2swz(s_kk[bb], lane, o, false);
  }
  __syncthreads();

  // ===================== point transformer (per batch) =====================
  for (int bb = 0; bb < GB; ++bb) {
    if (w < 6) {  // qkv = feats @ qkv_w (no bias); q row-major, k swizzled, v col-major
      v16bf a = load_Aswz(s_kk[bb], lane);  // feats(bb) from phase T0
      v8f c = bias_C(nullptr, 0, 1, lane);
      c = wmma_bf16(a, load_Bswz(ws + (C_QKV << 9), w, lane), c);
      if (w < 2)      store_Drm(s_q, (w & 1) * 16, lane, c);
      else if (w < 4) store_Dswz(s_k, (w & 1) * 16, lane, c, false);
      else            store_Dcm(s_v, (w & 1) * 16, lane, c);
    }
    __syncthreads();
    // pairwise rows j for each i; 2 i-tiles per wave
    for (int r = 0; r < 2; ++r) {
      const int i = w + r * 8;
      v16bf a1 = make_A(lane, [&](int m, int k) {
        return (k < 3) ? (s_coords[bb][i][k] - s_coords[bb][m][k]) : 0.f; });
      v8f rpe[2];
      mlp2_tile(a1, ws + (C_PTP1 << 9), p.ptp_b1, ws + (C_PTP2 << 9), p.ptp_b2,
                s_hscr[w], lane, rpe);
      store2swz(s_pe[w], lane, rpe, false);
      v16bf a2 = make_qkp(s_q + i * 32, s_k, s_pe[w], lane);
      v8f sim[2];
      mlp2_tile(a2, ws + (C_PTA1 << 9), p.pta_b1, ws + (C_PTA2 << 9), p.pta_b2,
                s_hscr[w], lane, sim);
      // per-channel softmax over j (rows split lane<->lane^16), then sum attn*(v+rpe)
      const int nlo = lane & 15, mb = (lane >> 4) << 3;
#pragma unroll
      for (int t = 0; t < 2; ++t) {
        int n = t * 16 + nlo;
        float mx = -3.4e38f;
#pragma unroll
        for (int v = 0; v < 8; ++v) mx = fmaxf(mx, sim[t][v]);
        mx = fmaxf(mx, xor16(mx));
        float ex[8], sum = 0.f;
#pragma unroll
        for (int v = 0; v < 8; ++v) { ex[v] = __expf(sim[t][v] - mx); sum += ex[v]; }
        sum += xor16(sum);
        v8bf vv8 = *(const v8bf*)(s_v + n * 16 + mb);
        float wsum = 0.f;
#pragma unroll
        for (int v = 0; v < 8; ++v) {
          int j = mb + v;
          wsum += (ex[v] / sum) * (bf2f(vv8[v]) + bf2f(s_pe[w][swz_idx(j, n)]));
        }
        wsum += xor16(wsum);
        if (lane < 16) s_fnew[swz_idx(i, n)] = f2bf(wsum);
      }
    }
    __syncthreads();
    if (w < 2) {  // step-invariant kk (swizzled, overwrites featAll[bb]), vv (col-major)
      v16bf a = load_Aswz(s_fnew, lane);
      v8f o[2];
      if (w == 0) {
        lin_tile(a, ws + (C_CAK << 9), p.cak_b, 32, lane, o);
        store2swz(s_kk[bb], lane, o, false);
      } else {
        lin_tile(a, ws + (C_CAV << 9), p.cav_b, 32, lane, o);
        store_Dcm(s_vv[bb], 0, lane, o[0]);
        store_Dcm(s_vv[bb], 16, lane, o[1]);
      }
    }
    __syncthreads();
  }

  // ============== march loop (+ final epilogue at s==STEPS) ==============
  for (int s = 0; s <= STEPS; ++s) {
    if (s < STEPS) {
      if (tid < GB) s_cumprev[tid] = s_cum[tid];
      __syncthreads();
      {  // rel_d per (batch, neighbor); emit local_depths
        int bb = tid >> 4, j = tid & 15;
        float cm = s_cum[bb];
        float rx = s_coords[bb][j][0] - s_query[bb][0] * cm;
        float ry = s_coords[bb][j][1] - s_query[bb][1] * cm;
        float rz = s_coords[bb][j][2] - s_query[bb][2] * cm;
        float d = sqrtf(rx * rx + ry * ry + rz * rz);
        s_reld[bb][j] = d;
        p.out_ld[((size_t)(b0 + bb) * STEPS + s) * 16 + j] = d;
      }
      __syncthreads();
    }
    if (w == 0) {  // qq = caq(feat_mlp(query*cum)); rows m = batch index
      v16bf a = make_A(lane, [&](int m, int k) {
        return (k < 3) ? s_query[m][k] * s_cum[m] : 0.f; });
      v8f o[2];
      lin_tile(a, ws + (C_FW1 << 9), p.fb1, 32, lane, o);  store2swz(s_hscr[0], lane, o, true);
      a = load_Aswz(s_hscr[0], lane);
      lin_tile(a, ws + (C_FW2 << 9), p.fb2, 32, lane, o);  store2swz(s_hscr[0], lane, o, true);
      a = load_Aswz(s_hscr[0], lane);
      lin_tile(a, ws + (C_FW3 << 9), p.fb3, 32, lane, o);  store2swz(s_hscr[0], lane, o, false);
      a = load_Aswz(s_hscr[0], lane);
      lin_tile(a, ws + (C_CAQ << 9), p.caq_b, 32, lane, o);
      store_Drm(s_qq, 0, lane, o[0]);
      store_Drm(s_qq, 16, lane, o[1]);
    }
    __syncthreads();
    // cross attention: 8 waves x 2 batches
    for (int r = 0; r < 2; ++r) {
      const int bb = w + r * 8;
      const float cm = (s == STEPS) ? s_cumprev[bb] : s_cum[bb];  // final reuses last rel
      v16bf a1 = make_A(lane, [&](int m, int k) {
        return (k < 3) ? (s_coords[bb][m][k] - s_query[bb][k] * cm) : 0.f; });
      v8f pe[2];
      mlp2_tile(a1, ws + (C_CAP1 << 9), p.cap_b1, ws + (C_CAP2 << 9), p.cap_b2,
                s_hscr[w], lane, pe);
      store2swz(s_pe[w], lane, pe, false);
      v16bf a2 = make_qkp(s_qq + bb * 32, s_kk[bb], s_pe[w], lane);
      v8f sim[2];
      mlp2_tile(a2, ws + (C_CAA1 << 9), p.caa_b1, ws + (C_CAA2 << 9), p.caa_b2,
                s_hscr[w], lane, sim);
      const int nlo = lane & 15, mb = (lane >> 4) << 3;
#pragma unroll
      for (int t = 0; t < 2; ++t) {
        int n = t * 16 + nlo;
        float mx = -3.4e38f;
#pragma unroll
        for (int v = 0; v < 8; ++v) mx = fmaxf(mx, sim[t][v]);
        mx = fmaxf(mx, xor16(mx));
        float ex[8], sum = 0.f;
#pragma unroll
        for (int v = 0; v < 8; ++v) { ex[v] = __expf(sim[t][v] - mx); sum += ex[v]; }
        sum += xor16(sum);
        v8bf vv8 = *(const v8bf*)(s_vv[bb] + n * 16 + mb);
        float wsum = 0.f;
#pragma unroll
        for (int v = 0; v < 8; ++v) {
          int j = mb + v;
          wsum += (ex[v] / sum) * (bf2f(vv8[v]) + bf2f(s_pe[w][swz_idx(j, n)]));
        }
        wsum += xor16(wsum);
        if (lane < 16) s_ca[bb * 32 + n] = wsum;
      }
    }
    __syncthreads();

    if (s < STEPS) {
      if (w == 0) {  // imp = MLP3(ca) -> [GB,3]
        v16bf a = make_A(lane, [&](int m, int k) { return s_ca[m * 32 + k]; });
        v8f o[2];
        lin_tile(a, ws + (C_IM1 << 9), p.im_b1, 32, lane, o); store2swz(s_hscr[0], lane, o, true);
        a = load_Aswz(s_hscr[0], lane);
        lin_tile(a, ws + (C_IM2 << 9), p.im_b2, 32, lane, o); store2swz(s_hscr[0], lane, o, true);
        a = load_Aswz(s_hscr[0], lane);
        lin_tile(a, ws + (C_IM3 << 9), p.im_b3, 3, lane, o);
        int n = lane & 15, mb = (lane >> 4) << 3;
        if (n < 3) {
#pragma unroll
          for (int v = 0; v < 8; ++v) s_imp[mb + v][n] = o[0][v];
        }
      }
      __syncthreads();
      {  // outputs using pre-update cum
        int bb = tid >> 4, j = tid & 15;
        float ix = s_imp[bb][0], iy = s_imp[bb][1], iz = s_imp[bb][2];
        float st = sqrtf(ix * ix + iy * iy + iz * iz);
        float cm = s_cum[bb];
        float rx = s_coords[bb][j][0] - s_query[bb][0] * cm;
        float ry = s_coords[bb][j][1] - s_query[bb][1] * cm;
        float rz = s_coords[bb][j][2] - s_query[bb][2] * cm;
        float d = s_reld[bb][j];
        p.out_cs[((size_t)(b0 + bb) * STEPS + s) * 16 + j] =
            (ix * rx + iy * ry + iz * rz) / (st * d);
        if (j == 0) {
          p.out_ms[(size_t)(b0 + bb) * STEPS + s] = st;
          size_t ip = ((size_t)(b0 + bb) * STEPS + s) * 3;
          p.out_ip[ip + 0] = s_query[bb][0] * cm + ix;
          p.out_ip[ip + 1] = s_query[bb][1] * cm + iy;
          p.out_ip[ip + 2] = s_query[bb][2] * cm + iz;
        }
      }
      __syncthreads();
      if (tid < GB) {
        float ix = s_imp[tid][0], iy = s_imp[tid][1], iz = s_imp[tid][2];
        s_cum[tid] += sqrtf(ix * ix + iy * iy + iz * iz);
      }
      __syncthreads();
    } else {
      if (w == 0) {  // eps = MLP3(concat(ca, query)): Cin=35 => two A k-chunks
        v16bf a0 = make_A(lane, [&](int m, int k) { return s_ca[m * 32 + k]; });
        v16bf aq = make_A(lane, [&](int m, int k) {
          return (k < 3) ? s_query[m][k] : 0.f; });
        v8f o[2];
#pragma unroll
        for (int t = 0; t < 2; ++t) {
          v8f c = bias_C(p.ep_b1, t * 16, 32, lane);
          c = wmma_bf16(a0, load_Bswz(ws + (C_EP1 << 9), 0 * 2 + t, lane), c);
          c = wmma_bf16(aq, load_Bswz(ws + (C_EP1 << 9), 1 * 2 + t, lane), c);
          o[t] = c;
        }
        store2swz(s_hscr[0], lane, o, true);
        v16bf a = load_Aswz(s_hscr[0], lane);
        lin_tile(a, ws + (C_EP2 << 9), p.ep_b2, 32, lane, o); store2swz(s_hscr[0], lane, o, true);
        a = load_Aswz(s_hscr[0], lane);
        lin_tile(a, ws + (C_EP3 << 9), p.ep_b3, 1, lane, o);
        int n = lane & 15, mb = (lane >> 4) << 3;
        if (n == 0) {
#pragma unroll
          for (int v = 0; v < 8; ++v) p.out_eps[b0 + mb + v] = o[0][v];
        }
      }
      if (tid < GB) p.out_cum[b0 + tid] = s_cum[tid];
    }
  }
}

extern "C" void kernel_launch(void* const* d_in, const int* in_sizes, int n_in,
                              void* d_out, int out_size, void* d_ws, size_t ws_size,
                              hipStream_t stream) {
  (void)n_in; (void)ws_size; (void)out_size;
  P p;
  p.coords = (const float*)d_in[0];  p.query  = (const float*)d_in[1];
  p.fw1 = (const float*)d_in[2];  p.fb1 = (const float*)d_in[3];
  p.fw2 = (const float*)d_in[4];  p.fb2 = (const float*)d_in[5];
  p.fw3 = (const float*)d_in[6];  p.fb3 = (const float*)d_in[7];
  p.qkv_w = (const float*)d_in[8];
  p.ptp_w1 = (const float*)d_in[9];  p.ptp_b1 = (const float*)d_in[10];
  p.ptp_w2 = (const float*)d_in[11]; p.ptp_b2 = (const float*)d_in[12];
  p.pta_w1 = (const float*)d_in[13]; p.pta_b1 = (const float*)d_in[14];
  p.pta_w2 = (const float*)d_in[15]; p.pta_b2 = (const float*)d_in[16];
  p.caq_w = (const float*)d_in[17];  p.caq_b = (const float*)d_in[18];
  p.cak_w = (const float*)d_in[19];  p.cak_b = (const float*)d_in[20];
  p.cav_w = (const float*)d_in[21];  p.cav_b = (const float*)d_in[22];
  p.cap_w1 = (const float*)d_in[23]; p.cap_b1 = (const float*)d_in[24];
  p.cap_w2 = (const float*)d_in[25]; p.cap_b2 = (const float*)d_in[26];
  p.caa_w1 = (const float*)d_in[27]; p.caa_b1 = (const float*)d_in[28];
  p.caa_w2 = (const float*)d_in[29]; p.caa_b2 = (const float*)d_in[30];
  p.im_w1 = (const float*)d_in[31];  p.im_b1 = (const float*)d_in[32];
  p.im_w2 = (const float*)d_in[33];  p.im_b2 = (const float*)d_in[34];
  p.im_w3 = (const float*)d_in[35];  p.im_b3 = (const float*)d_in[36];
  p.ep_w1 = (const float*)d_in[37];  p.ep_b1 = (const float*)d_in[38];
  p.ep_w2 = (const float*)d_in[39];  p.ep_b2 = (const float*)d_in[40];
  p.ep_w3 = (const float*)d_in[41];  p.ep_b3 = (const float*)d_in[42];

  const int B = in_sizes[1] / 3;
  float* out = (float*)d_out;
  p.out_ld  = out;
  p.out_ms  = p.out_ld + (size_t)B * STEPS * 16;
  p.out_cs  = p.out_ms + (size_t)B * STEPS;
  p.out_ip  = p.out_cs + (size_t)B * STEPS * 16;
  p.out_cum = p.out_ip + (size_t)B * STEPS * 3;
  p.out_eps = p.out_cum + B;

  __bf16* ws = (__bf16*)d_ws;
  swz_weights<<<21, 256, 0, stream>>>(p, ws);
  puray_cdna5_kernel<<<B / GB, 256, 0, stream>>>(p, ws);
}
